// _NonLocalBlockND_9010841387873
// MI455X (gfx1250) — matrix operand order repaired and checked by reading, not compiled, not run on hardware
//
#include <hip/hip_runtime.h>

typedef __attribute__((ext_vector_type(16))) __bf16 v16bf;
typedef __attribute__((ext_vector_type(8)))  __bf16 v8bf;
typedef __attribute__((ext_vector_type(8)))  float  v8f;
typedef __attribute__((ext_vector_type(4)))  unsigned int v4u;
typedef __attribute__((ext_vector_type(8)))  int  v8i;
typedef __attribute__((ext_vector_type(4)))  int  v4i;

#define B_N      16
#define C_IN     64
#define C_INT    32
#define HW       4096   // 64*64
#define NKV      1024   // 32*32
#define LROW     40     // LDS row stride in bf16 elems (64B data + 16B TDM pad)

#define WMMA_BF16(A, B, C) \
    __builtin_amdgcn_wmma_f32_16x16x32_bf16(false, (A), false, (B), (short)0, (C), false, false)

// 16-lane xor-tree reduction stage via wave32 DPP row ops (no DS traffic).
#define DPP_MAX(r, ctrl) \
    r = fmaxf(r, __int_as_float(__builtin_amdgcn_update_dpp(0, __float_as_int(r), (ctrl), 0xF, 0xF, true)))

// ---------------------------------------------------------------------------
// TDM: DMA a 2D bf16 tile (tile0 x tile1, row stride stride0 elems) from
// global memory into LDS, padding each 64B row by 16B (rows land 80B apart).
// D# layout per CDNA5 ISA ch.8; groups 2/3 zero => 2D tensor.
// ---------------------------------------------------------------------------
__device__ __forceinline__ void tdm_load_tile(unsigned lds_off, const void* g,
                                              unsigned tile0, unsigned tile1,
                                              unsigned td0, unsigned td1,
                                              unsigned long long stride0)
{
    const unsigned long long ga = (unsigned long long)g;
    v4u g0;
    g0[0] = 1u;                                             // count=1 (valid user D#)
    g0[1] = lds_off;                                        // lds_addr (bytes)
    g0[2] = (unsigned)(ga & 0xFFFFFFFFu);                   // global_addr[31:0]
    g0[3] = (unsigned)((ga >> 32) & 0x01FFFFFFu) | 0x80000000u;  // addr[56:32] | type=2
    v8i g1;
    g1[0] = (int)0x06D10000u;   // data_size=2B | pad_enable | interval=64B | amount=16B
    g1[1] = (int)((td0 & 0xFFFFu) << 16);
    g1[2] = (int)((td0 >> 16) | ((td1 & 0xFFFFu) << 16));
    g1[3] = (int)((td1 >> 16) | (tile0 << 16));
    g1[4] = (int)(tile1 & 0xFFFFu);                         // tile_dim1 (tile_dim2=0)
    g1[5] = (int)(unsigned)(stride0 & 0xFFFFFFFFu);         // dim0 stride lo
    g1[6] = (int)(unsigned)((stride0 >> 32) & 0xFFFFu);     // dim0 stride hi
    g1[7] = 0;
    const v4i z4 = {};
    const v8i z8 = {};
    __builtin_amdgcn_tensor_load_to_lds(g0, g1, z4, z4, z8, 0);
}

// ---------------------------------------------------------------------------
// Kernel 0: transpose + convert x [B][64][4096] fp32 -> xb [B][4096][64] bf16
// ---------------------------------------------------------------------------
__global__ __launch_bounds__(256) void convert_x_kernel(
    const float* __restrict__ x, __bf16* __restrict__ xb)
{
    __shared__ float tile[64][65];
    const int b  = blockIdx.y;
    const int n0 = blockIdx.x * 64;
    const int tn = threadIdx.x & 63, tc = threadIdx.x >> 6;  // tc in 0..3
#pragma unroll
    for (int c = tc; c < 64; c += 4)
        tile[c][tn] = x[((size_t)b * C_IN + c) * HW + n0 + tn];
    __syncthreads();
#pragma unroll
    for (int nn = tc; nn < 64; nn += 4)
        xb[((size_t)b * HW + n0 + nn) * C_IN + tn] = (__bf16)tile[tn][nn];
}

// B-operand (32x16 bf16) column segment from fp32 weight matrix w[o][ldw].
__device__ __forceinline__ v16bf load_wB(const float* __restrict__ w, int ldw,
                                         int o, int kbase)
{
    const float* wr = w + (size_t)o * ldw + kbase;
    v16bf r;
#pragma unroll
    for (int i = 0; i < 16; ++i) r[i] = (__bf16)wr[i];
    return r;
}

// A-operand 32-K slice (contiguous k = 16h + s order) from a bf16 row.
__device__ __forceinline__ v16bf load_aA(const __bf16* __restrict__ row, int koff)
{
    const v8bf lo = *(const v8bf*)(row + koff);
    const v8bf hi = *(const v8bf*)(row + koff + 8);
    v16bf r;
#pragma unroll
    for (int i = 0; i < 8; ++i) { r[i] = lo[i]; r[i + 8] = hi[i]; }
    return r;
}

// B-operand: 16 contiguous bf16 from LDS as two 16B ds loads (rows 16B-aligned).
__device__ __forceinline__ v16bf load_bLDS(const __bf16* p)
{
    const v8bf lo = *(const v8bf*)p;
    const v8bf hi = *(const v8bf*)(p + 8);
    v16bf r;
#pragma unroll
    for (int i = 0; i < 8; ++i) { r[i] = lo[i]; r[i + 8] = hi[i]; }
    return r;
}

// ---------------------------------------------------------------------------
// Kernel 1: theta projection on WMMA -> Q bf16 [B][4096][32]
// ---------------------------------------------------------------------------
__global__ __launch_bounds__(128) void proj_theta_wmma(
    const __bf16* __restrict__ xb, const float* __restrict__ w,
    const float* __restrict__ bias, __bf16* __restrict__ Q)
{
    const int b     = blockIdx.y;
    const int wave  = threadIdx.x >> 5, lane = threadIdx.x & 31;
    const int lhalf = lane >> 4, l16 = lane & 15, koff = lhalf * 16;
    const int n0    = blockIdx.x * 64 + wave * 16;

    v16bf wb[2][2];
#pragma unroll
    for (int ks = 0; ks < 2; ++ks)
#pragma unroll
        for (int h = 0; h < 2; ++h)
            wb[ks][h] = load_wB(w, C_IN, h * 16 + l16, ks * 32 + lhalf * 16);

    const float bv0 = bias[l16], bv1 = bias[16 + l16];
    v8f acc0, acc1;
#pragma unroll
    for (int j = 0; j < 8; ++j) { acc0[j] = bv0; acc1[j] = bv1; }

    const __bf16* xrow = xb + (size_t)(b * HW + n0 + l16) * C_IN;
#pragma unroll
    for (int ks = 0; ks < 2; ++ks) {
        const v16bf xa = load_aA(xrow, ks * 32 + koff);
        acc0 = WMMA_BF16(xa, wb[ks][0], acc0);
        acc1 = WMMA_BF16(xa, wb[ks][1], acc1);
    }
#pragma unroll
    for (int j = 0; j < 8; ++j) {
        const int row = j + 8 * lhalf;
        __bf16* qp = Q + (size_t)(b * HW + n0 + row) * C_INT;
        qp[l16]      = (__bf16)acc0[j];
        qp[16 + l16] = (__bf16)acc1[j];
    }
}

// ---------------------------------------------------------------------------
// Kernel 2: phi + origin projections on WMMA, fused 2x2 maxpool
//   -> K bf16 [B][1024][32], Vt bf16 [B][32][1024]
// ---------------------------------------------------------------------------
__global__ __launch_bounds__(256) void proj_pooled_wmma(
    const __bf16* __restrict__ xb,
    const float* __restrict__ w_phi, const float* __restrict__ b_phi,
    const float* __restrict__ w_org, const float* __restrict__ b_org,
    __bf16* __restrict__ K, __bf16* __restrict__ Vt)
{
    __shared__ float phis[2 * 64 * 32];
    __shared__ float orgs[2 * 64 * 32];
    const int b     = blockIdx.y;
    const int wave  = threadIdx.x >> 5, lane = threadIdx.x & 31;
    const int lhalf = lane >> 4, l16 = lane & 15, koff = lhalf * 16;
    const int r     = wave >> 2;                 // sub-row 0/1
    const int h     = blockIdx.x * 2 + r;        // image row
    const int wseg  = (wave & 3) * 16;
    const int n0    = h * 64 + wseg;

    v16bf wbp[2][2], wbo[2][2];
#pragma unroll
    for (int ks = 0; ks < 2; ++ks)
#pragma unroll
        for (int hh = 0; hh < 2; ++hh) {
            wbp[ks][hh] = load_wB(w_phi, C_IN, hh * 16 + l16, ks * 32 + lhalf * 16);
            wbo[ks][hh] = load_wB(w_org, C_IN, hh * 16 + l16, ks * 32 + lhalf * 16);
        }

    const float bp0 = b_phi[l16], bp1 = b_phi[16 + l16];
    const float bo0 = b_org[l16], bo1 = b_org[16 + l16];
    v8f ap0, ap1, ao0, ao1;
#pragma unroll
    for (int j = 0; j < 8; ++j) { ap0[j] = bp0; ap1[j] = bp1; ao0[j] = bo0; ao1[j] = bo1; }

    const __bf16* xrow = xb + (size_t)(b * HW + n0 + l16) * C_IN;
#pragma unroll
    for (int ks = 0; ks < 2; ++ks) {
        const v16bf xa = load_aA(xrow, ks * 32 + koff);
        ap0 = WMMA_BF16(xa, wbp[ks][0], ap0);
        ap1 = WMMA_BF16(xa, wbp[ks][1], ap1);
        ao0 = WMMA_BF16(xa, wbo[ks][0], ao0);
        ao1 = WMMA_BF16(xa, wbo[ks][1], ao1);
    }
#pragma unroll
    for (int j = 0; j < 8; ++j) {
        const int wc = wseg + j + 8 * lhalf;     // w-coordinate 0..63
        float* pp = phis + ((size_t)r * 64 + wc) * 32;
        float* op = orgs + ((size_t)r * 64 + wc) * 32;
        pp[l16] = ap0[j]; pp[16 + l16] = ap1[j];
        op[l16] = ao0[j]; op[16 + l16] = ao1[j];
    }
    __syncthreads();

    const int o   = threadIdx.x & 31;
    const int wp0 = threadIdx.x >> 5;            // 0..7
#pragma unroll
    for (int wp = wp0; wp < 32; wp += 8) {
        const int np = blockIdx.x * 32 + wp;
        const int i00 = (2 * wp) * 32 + o, i01 = (2 * wp + 1) * 32 + o;
        const float pv = fmaxf(fmaxf(phis[i00], phis[i01]),
                               fmaxf(phis[64 * 32 + i00], phis[64 * 32 + i01]));
        const float vv = fmaxf(fmaxf(orgs[i00], orgs[i01]),
                               fmaxf(orgs[64 * 32 + i00], orgs[64 * 32 + i01]));
        K[((size_t)b * NKV + np) * C_INT + o]  = (__bf16)pv;
        Vt[((size_t)b * C_INT + o) * NKV + np] = (__bf16)vv;
    }
}

// ---------------------------------------------------------------------------
// Kernel 3: flash attention. 4 waves/block, 16 queries/wave.
//   K/V chunks DMA'd by the TDM into double-buffered LDS one iteration ahead
//   (wave 0 issues tensor_load_to_lds, s_wait_tensorcnt 2 keeps the prefetch
//   in flight). S via 2 WMMAs, row-max via DPP xor-tree, row-sum via WMMA
//   against ones, P restriped D->A through per-wave LDS tile, P@V via 2 WMMAs.
//   -> Y bf16 [B][4096][32]
// ---------------------------------------------------------------------------
__global__ __launch_bounds__(128) void flash_attn_kernel(
    const __bf16* __restrict__ Q, const __bf16* __restrict__ Kmat,
    const __bf16* __restrict__ Vt, __bf16* __restrict__ Y)
{
    __shared__ alignas(32) __bf16 kv[2][2][32 * LROW];   // [buf][K/V][32 rows x 80B]
    __shared__ float plds[4][16 * 33];

    const int b     = blockIdx.y;
    const int wave  = threadIdx.x >> 5, lane = threadIdx.x & 31;
    const int lhalf = lane >> 4, l16 = lane & 15, koff = lhalf * 16;
    const int q0    = (blockIdx.x * 4 + wave) * 16;

    const __bf16* qrow = Q + (size_t)(b * HW + q0 + l16) * C_INT;
    const v16bf qa = load_aA(qrow, koff);

    v16bf onesb;
#pragma unroll
    for (int i = 0; i < 16; ++i) onesb[i] = (__bf16)1.0f;

    float m[8];
    v8f lacc = {}, acc0 = {}, acc1 = {};
#pragma unroll
    for (int j = 0; j < 8; ++j) m[j] = -__builtin_inff();
    const v8f zeroc = {};

    const __bf16* Kb = Kmat + (size_t)b * NKV * C_INT;
    const __bf16* Vb = Vt   + (size_t)b * C_INT * NKV;

    // prime chunk 0 into buffer 0 (one DMA pair per block, wave 0 only)
    if (threadIdx.x < 32) {
        tdm_load_tile((unsigned)(size_t)&kv[0][0][0], Kb, 32, 32, 32, 32, 32);
        tdm_load_tile((unsigned)(size_t)&kv[0][1][0], Vb, 32, 32, 32, 32, NKV);
    }

    for (int ci = 0; ci < NKV / 32; ++ci) {
        const int nc = (ci + 1) & 31;            // wraps harmlessly on last iter
        const int nb = (ci + 1) & 1;
        if (threadIdx.x < 32) {
            tdm_load_tile((unsigned)(size_t)&kv[nb][0][0], Kb + (size_t)nc * 32 * C_INT,
                          32, 32, 32, 32, 32);
            tdm_load_tile((unsigned)(size_t)&kv[nb][1][0], Vb + (size_t)nc * 32,
                          32, 32, 32, 32, NKV);
            __builtin_amdgcn_s_wait_tensorcnt(2);   // chunk ci resident; prefetch in flight
        }
        __syncthreads();

        const __bf16* kt = kv[ci & 1][0];
        const __bf16* vt = kv[ci & 1][1];
        const v16bf kb0 = load_bLDS(kt + l16 * LROW + lhalf * 16);
        const v16bf kb1 = load_bLDS(kt + (16 + l16) * LROW + lhalf * 16);

        v8f s0 = WMMA_BF16(qa, kb0, zeroc);
        v8f s1 = WMMA_BF16(qa, kb1, zeroc);

        float p0[8], p1[8];
#pragma unroll
        for (int j = 0; j < 8; ++j) {
            float r = fmaxf(s0[j], s1[j]);
            DPP_MAX(r, 0xB1);   // xor 1 (quad_perm 1,0,3,2)
            DPP_MAX(r, 0x4E);   // xor 2 (quad_perm 2,3,0,1)
            DPP_MAX(r, 0x141);  // row_half_mirror == xor4-equivalent
            DPP_MAX(r, 0x140);  // row_mirror      == xor8-equivalent
            const float mn = fmaxf(m[j], r);
            const float sc = __expf(m[j] - mn);
            p0[j] = __expf(s0[j] - mn);
            p1[j] = __expf(s1[j] - mn);
            m[j] = mn;
            lacc[j] *= sc;
            acc0[j] *= sc;
            acc1[j] *= sc;
        }

        // restripe P from C/D layout to A layout via per-wave LDS tile
        float* pl = plds[wave];
#pragma unroll
        for (int j = 0; j < 8; ++j) {
            const int row = j + 8 * lhalf;
            pl[row * 33 + l16]      = p0[j];
            pl[row * 33 + 16 + l16] = p1[j];
        }
        asm volatile("" ::: "memory");   // same-wave DS ops are in-order
        const float* prow = pl + l16 * 33;
        v16bf pa;
#pragma unroll
        for (int i = 0; i < 8; ++i) {
            pa[i]     = (__bf16)prow[koff + i];
            pa[i + 8] = (__bf16)prow[koff + 8 + i];
        }
        asm volatile("" ::: "memory");

        // row-sum of P via WMMA against ones: lands lane-local in D layout
        lacc = WMMA_BF16(pa, onesb, lacc);

        const v16bf vb0 = load_bLDS(vt + l16 * LROW + lhalf * 16);
        const v16bf vb1 = load_bLDS(vt + (16 + l16) * LROW + lhalf * 16);
        acc0 = WMMA_BF16(pa, vb0, acc0);
        acc1 = WMMA_BF16(pa, vb1, acc1);

        __syncthreads();   // all waves done with buf (ci&1) before it is re-filled
    }

#pragma unroll
    for (int j = 0; j < 8; ++j) {
        const int row = j + 8 * lhalf;
        const float inv = 1.0f / lacc[j];
        __bf16* yrow = Y + (size_t)(b * HW + q0 + row) * C_INT;
        yrow[l16]      = (__bf16)(acc0[j] * inv);
        yrow[16 + l16] = (__bf16)(acc1[j] * inv);
    }
}

// ---------------------------------------------------------------------------
// Kernel 4: final conv on WMMA: z = y @ w_W^T + b_W + x (fp32 out, residual)
// ---------------------------------------------------------------------------
__global__ __launch_bounds__(128) void final_conv_wmma(
    const __bf16* __restrict__ Y, const float* __restrict__ wW,
    const float* __restrict__ bW, const float* __restrict__ x,
    float* __restrict__ z)
{
    __shared__ float zt[64 * 65];
    const int b     = blockIdx.y;
    const int wave  = threadIdx.x >> 5, lane = threadIdx.x & 31;
    const int lhalf = lane >> 4, l16 = lane & 15, koff = lhalf * 16;
    const int n0    = blockIdx.x * 64;
    const int nw    = n0 + wave * 16;

    v16bf wb[4];
    v8f acc[4];
#pragma unroll
    for (int h = 0; h < 4; ++h) {
        wb[h] = load_wB(wW, C_INT, h * 16 + l16, lhalf * 16);
        const float bv = bW[h * 16 + l16];
#pragma unroll
        for (int j = 0; j < 8; ++j) acc[h][j] = bv;
    }

    const __bf16* yrow = Y + (size_t)(b * HW + nw + l16) * C_INT;
    const v16bf ya = load_aA(yrow, koff);
#pragma unroll
    for (int h = 0; h < 4; ++h) acc[h] = WMMA_BF16(ya, wb[h], acc[h]);

#pragma unroll
    for (int h = 0; h < 4; ++h)
#pragma unroll
        for (int j = 0; j < 8; ++j) {
            const int r = wave * 16 + j + 8 * lhalf;
            zt[r * 65 + h * 16 + l16] = acc[h][j];
        }
    __syncthreads();

    const int p  = threadIdx.x & 63;
    const int cb = (threadIdx.x >> 6) * 32;      // 0 or 32
#pragma unroll
    for (int co = cb; co < cb + 32; ++co) {
        const size_t gi = (size_t)(b * C_IN + co) * HW + n0 + p;
        z[gi] = zt[p * 65 + co] + x[gi];
    }
}

// ---------------------------------------------------------------------------
extern "C" void kernel_launch(void* const* d_in, const int* in_sizes, int n_in,
                              void* d_out, int out_size, void* d_ws, size_t ws_size,
                              hipStream_t stream) {
    const float* x        = (const float*)d_in[0];
    const float* w_origin = (const float*)d_in[1];
    const float* b_origin = (const float*)d_in[2];
    const float* w_theta  = (const float*)d_in[3];
    const float* b_theta  = (const float*)d_in[4];
    const float* w_phi    = (const float*)d_in[5];
    const float* b_phi    = (const float*)d_in[6];
    const float* w_W      = (const float*)d_in[7];
    const float* b_W      = (const float*)d_in[8];
    float* z = (float*)d_out;

    char* ws = (char*)d_ws;
    __bf16* Q  = (__bf16*)(ws);                        // 16*4096*32*2 = 4 MB
    __bf16* K  = (__bf16*)(ws + ((size_t)4  << 20));   // 1 MB
    __bf16* Vt = (__bf16*)(ws + ((size_t)5  << 20));   // 1 MB
    __bf16* Yb = (__bf16*)(ws + ((size_t)6  << 20));   // 4 MB
    __bf16* xb = (__bf16*)(ws + ((size_t)10 << 20));   // 16*4096*64*2 = 8 MB

    convert_x_kernel <<<dim3(64, B_N), 256, 0, stream>>>(x, xb);
    proj_theta_wmma  <<<dim3(64, B_N), 128, 0, stream>>>(xb, w_theta, b_theta, Q);
    proj_pooled_wmma <<<dim3(32, B_N), 256, 0, stream>>>(xb, w_phi, b_phi,
                                                         w_origin, b_origin, K, Vt);
    flash_attn_kernel<<<dim3(64, B_N), 128, 0, stream>>>(Q, K, Vt, Yb);
    final_conv_wmma  <<<dim3(64, B_N), 128, 0, stream>>>(Yb, w_W, b_W, x, z);
}
